// Proposed_network_35141422416467
// MI455X (gfx1250) — matrix operand order repaired and checked by reading, not compiled, not run on hardware
//
#include <hip/hip_runtime.h>
#include <hip/hip_bf16.h>
#include <math.h>

// ---------------------------------------------------------------------------
// CDNA5 (gfx1250, wave32) implementation of the RandLA-Net-style reference.
// All 1x1 convs / linear layers go through one fused WMMA GEMM kernel:
//   Y[b,m,n] = act( gamma[m] * sum_k W[m,k]*X[b,k,n] + beta[m] (+ R[b,m,n]) )
// using v_wmma_f32_16x16x32_f16 (f16 inputs, f32 accumulation), with
// fragment-layout LDS tiles (ds_load_b128 fragments), batched fragment loads
// feeding 4 back-to-back WMMAs, and a double-buffered staging pipeline.
// ---------------------------------------------------------------------------

typedef __attribute__((ext_vector_type(16))) _Float16 v16h;
typedef __attribute__((ext_vector_type(8)))  _Float16 v8h;
typedef __attribute__((ext_vector_type(8)))  float    v8f;

#define NL 4
#define KNN 16

// activation codes
#define ACT_NONE   0
#define ACT_RELU   1
#define ACT_SIGM   2
#define ACT_LEAK2  3   // leaky 0.2
#define ACT_LEAK01 4   // leaky 0.01

#define BM 64
#define BN 128
#define BK 32

// ---------------------------------------------------------------------------
// Fused WMMA GEMM: 256 threads = 8 waves; block tile 64(M) x 128(N), K step 32.
// Wave w: row subtile (w>>1), col range (w&1)*64 .. +64 (4 accumulators).
// LDS layouts are fragment-major:  As[m][k],  Bs[n][k]  (X transposed on the
// way in), so every WMMA fragment is two contiguous 16B LDS reads.
// ---------------------------------------------------------------------------
__global__ __launch_bounds__(256)
void gemm_bn_act_kernel(const float* __restrict__ X, const float* __restrict__ W,
                        const float* __restrict__ gamma, const float* __restrict__ beta,
                        const float* __restrict__ R, float* __restrict__ Y,
                        int Cin, int Cout, int N, int act)
{
    __shared__ __align__(16) _Float16 As[2][BM * BK];   // [m][k]
    __shared__ __align__(16) _Float16 Bs[2][BN * BK];   // [n][k]

    const int tid  = threadIdx.x;
    const int lane = tid & 31;
    const int wv   = tid >> 5;          // 0..7
    const int h    = lane >> 4;         // lane half
    const int l    = lane & 15;
    const int mi   = wv >> 1;           // row subtile 0..3
    const int nc0  = (wv & 1) * 64;     // col base; wave covers 4 subtiles of 16

    const int bM = blockIdx.y * BM;
    const int bN = blockIdx.x * BN;
    const int b  = blockIdx.z;
    const size_t xoff = (size_t)b * Cin * N;
    const size_t yoff = (size_t)b * Cout * N;

    // ---- staging helpers (block-uniform fast/slow path selection) ----
    auto stageA = [&](int k0, int p) {
        if (bM + BM <= Cout && k0 + BK <= Cin) {
            // 64 m x 4 k-groups of 8 = 256 items, one per thread
            int m  = tid & 63;
            int kg = (tid >> 6) * 8;
            const float* src = W + (size_t)(bM + m) * Cin + (k0 + kg);
            v8h d;
#pragma unroll
            for (int j = 0; j < 8; ++j) d[j] = (_Float16)src[j];
            *(v8h*)&As[p][m * BK + kg] = d;
        } else {
            // lane index walks k (memory-contiguous) -> coalesced
            for (int e = tid; e < BM * BK; e += 256) {
                int m = e >> 5, k = e & 31;
                float v = (bM + m < Cout && k0 + k < Cin)
                              ? W[(size_t)(bM + m) * Cin + (k0 + k)] : 0.f;
                As[p][m * BK + k] = (_Float16)v;
            }
        }
    };
    auto stageB = [&](int k0, int p) {
        if (bN + BN <= N && k0 + BK <= Cin) {
            // 128 n x 4 k-groups of 8 = 512 items -> 2 per thread.
            // n = tid&127 keeps global loads lane-contiguous (fully coalesced).
#pragma unroll
            for (int it = 0; it < 2; ++it) {
                int n  = tid & 127;
                int kg = ((tid >> 7) + it * 2) * 8;
                const float* src = X + xoff + (size_t)(k0 + kg) * N + bN + n;
                v8h d;
#pragma unroll
                for (int j = 0; j < 8; ++j) d[j] = (_Float16)src[(size_t)j * N];
                *(v8h*)&Bs[p][n * BK + kg] = d;
            }
        } else {
            // lane index walks n (memory-contiguous) -> coalesced
            for (int e = tid; e < BN * BK; e += 256) {
                int k = e >> 7, n = e & 127;
                float v = (bN + n < N && k0 + k < Cin)
                              ? X[xoff + (size_t)(k0 + k) * N + bN + n] : 0.f;
                Bs[p][n * BK + k] = (_Float16)v;
            }
        }
    };

    v8f acc[4] = {};
    stageA(0, 0);
    stageB(0, 0);
    int p = 0;
    for (int k0 = 0; k0 < Cin; k0 += BK) {
        __syncthreads();
        if (k0 + BK < Cin) { stageA(k0 + BK, p ^ 1); stageB(k0 + BK, p ^ 1); }

        // Batch-load all fragments first so one dscnt wait covers all of them,
        // then issue the 4 WMMAs back-to-back.
        // A fragment (ISA 7.12.2, 16-bit A 16x32): e<8 -> k=h*8+e ; e>=8 -> 16+h*8+(e-8)
        const _Float16* ar = &As[p][(mi * 16 + l) * BK];
        v8h alo = *(const v8h*)(ar + h * 8);
        v8h ahi = *(const v8h*)(ar + 16 + h * 8);
        v16h a = __builtin_shufflevector(alo, ahi, 0, 1, 2, 3, 4, 5, 6, 7,
                                         8, 9, 10, 11, 12, 13, 14, 15);
        // B fragments (16-bit B 32x16): e<8 -> k=h*8+e ; e>=8 -> 16+h*8+(e-8), n = lane&15
        v16h bf[4];
#pragma unroll
        for (int t = 0; t < 4; ++t) {
            const _Float16* br = &Bs[p][(nc0 + t * 16 + l) * BK + h * 8];
            v8h blo = *(const v8h*)(br);
            v8h bhi = *(const v8h*)(br + 16);
            bf[t] = __builtin_shufflevector(blo, bhi, 0, 1, 2, 3, 4, 5, 6, 7,
                                            8, 9, 10, 11, 12, 13, 14, 15);
        }
#pragma unroll
        for (int t = 0; t < 4; ++t)
            acc[t] = __builtin_amdgcn_wmma_f32_16x16x32_f16(false, a, false, bf[t],
                                                            (short)0, acc[t], false, false);
        p ^= 1;
    }

    // Epilogue: D layout -- VGPR r holds m = r + 8*h, col = lane&15
#pragma unroll
    for (int r = 0; r < 8; ++r) {
        int m = bM + mi * 16 + r + 8 * h;
        if (m >= Cout) continue;
        float g  = gamma ? gamma[m] : 1.f;
        float be = beta  ? beta[m]  : 0.f;
#pragma unroll
        for (int t = 0; t < 4; ++t) {
            int col = bN + nc0 + t * 16 + l;
            if (col >= N) continue;
            float v = acc[t][r] * g + be;
            size_t oi = yoff + (size_t)m * N + col;
            if (R) v += R[oi];
            if      (act == ACT_RELU)   v = fmaxf(v, 0.f);
            else if (act == ACT_SIGM)   v = 1.f / (1.f + __expf(-v));
            else if (act == ACT_LEAK2)  v = v > 0.f ? v : 0.2f * v;
            else if (act == ACT_LEAK01) v = v > 0.f ? v : 0.01f * v;
            Y[oi] = v;
        }
    }
}

// ---------------------------------------------------------------------------
// fc0: [B,N,6] -> [B,8,N], y = leaky_0.01((x @ W^T + b) * g + be)
// ---------------------------------------------------------------------------
__global__ void fc0_kernel(const float* __restrict__ feat, const float* __restrict__ w,
                           const float* __restrict__ bias, const float* __restrict__ g,
                           const float* __restrict__ be, float* __restrict__ out, int N)
{
    int t = blockIdx.x * blockDim.x + threadIdx.x;
    if (t >= 2 * N) return;
    int b = t / N, n = t % N;
    const float* x = feat + ((size_t)b * N + n) * 6;
    float xv[6];
#pragma unroll
    for (int c = 0; c < 6; ++c) xv[c] = x[c];
#pragma unroll
    for (int o = 0; o < 8; ++o) {
        float s = bias[o];
#pragma unroll
        for (int c = 0; c < 6; ++c) s += w[o * 6 + c] * xv[c];
        s = s * g[o] + be[o];
        s = s > 0.f ? s : 0.01f * s;
        out[(size_t)b * 8 * N + (size_t)o * N + n] = s;
    }
}

// ---------------------------------------------------------------------------
// Local feature aggregation: per point, relative-pos encoding MLP (10->Chalf)
// + gathered neighbor features, max over K=16.  out: [B, 2*Chalf, N]
// ---------------------------------------------------------------------------
__global__ __launch_bounds__(64)
void lfa_kernel(const float* __restrict__ xyz, const float* __restrict__ f,
                const int* __restrict__ neigh,
                const float* __restrict__ pe_w, const float* __restrict__ pe_g,
                const float* __restrict__ pe_b,
                float* __restrict__ out, int Chalf, int N)
{
    __shared__ float pe10[KNN][10];
    __shared__ int   nidx[KNN];
    const int b = blockIdx.y;
    const int n = blockIdx.x;
    const int tid = threadIdx.x;
    const float* xb = xyz + (size_t)b * N * 3;

    if (tid < KNN) {
        int idx = neigh[((size_t)b * N + n) * KNN + tid];
        nidx[tid] = idx;
        float cx = xb[(size_t)n * 3 + 0], cy = xb[(size_t)n * 3 + 1], cz = xb[(size_t)n * 3 + 2];
        float nx = xb[(size_t)idx * 3 + 0], ny = xb[(size_t)idx * 3 + 1], nz = xb[(size_t)idx * 3 + 2];
        float rx = nx - cx, ry = ny - cy, rz = nz - cz;
        float d = sqrtf(rx * rx + ry * ry + rz * rz + 1e-12f);
        pe10[tid][0] = cx; pe10[tid][1] = cy; pe10[tid][2] = cz;
        pe10[tid][3] = nx; pe10[tid][4] = ny; pe10[tid][5] = nz;
        pe10[tid][6] = rx; pe10[tid][7] = ry; pe10[tid][8] = rz;
        pe10[tid][9] = d;
    }
    __syncthreads();

    const float* fb = f + (size_t)b * Chalf * N;
    float* ob = out + (size_t)b * (2 * Chalf) * N;
    for (int co = tid; co < Chalf; co += 64) {
        const float* wr = pe_w + (size_t)co * 10;
        float pg = pe_g[co], pb = pe_b[co];
        float pm = -INFINITY, nm = -INFINITY;
#pragma unroll
        for (int k = 0; k < KNN; ++k) {
            float s = 0.f;
#pragma unroll
            for (int c = 0; c < 10; ++c) s += wr[c] * pe10[k][c];
            s = fmaxf(s * pg + pb, 0.f);
            pm = fmaxf(pm, s);
            nm = fmaxf(nm, fb[(size_t)co * N + nidx[k]]);
        }
        ob[(size_t)co * N + n]           = nm;  // neighbor-feature half
        ob[(size_t)(Chalf + co) * N + n] = pm;  // pos-encoding half
    }
}

// out[b,c,m] = max_k f[b,c,idx[b,m,k]]   (K = 16)
__global__ void pool_max_kernel(const float* __restrict__ f, const int* __restrict__ idx,
                                float* __restrict__ out, int C, int Nsrc, int M)
{
    int t = blockIdx.x * blockDim.x + threadIdx.x;
    int total = 2 * C * M;
    if (t >= total) return;
    int b = t / (C * M);
    int r = t % (C * M);
    int c = r / M, m = r % M;
    const int* ip = idx + ((size_t)b * M + m) * KNN;
    const float* fp = f + ((size_t)b * C + c) * Nsrc;
    float v = -INFINITY;
#pragma unroll
    for (int k = 0; k < KNN; ++k) v = fmaxf(v, fp[ip[k]]);
    out[((size_t)b * C + c) * M + m] = v;
}

// out[b,c,m] = f[b,c,idx[b,m]]
__global__ void gather_kernel(const float* __restrict__ f, const int* __restrict__ idx,
                              float* __restrict__ out, int C, int Nsrc, int M)
{
    int t = blockIdx.x * blockDim.x + threadIdx.x;
    int total = 2 * C * M;
    if (t >= total) return;
    int b = t / (C * M);
    int r = t % (C * M);
    int c = r / M, m = r % M;
    out[((size_t)b * C + c) * M + m] = f[((size_t)b * C + c) * Nsrc + idx[(size_t)b * M + m]];
}

// a[i] = relu(a[i] + x[i])
__global__ void add_relu_kernel(float* __restrict__ a, const float* __restrict__ x, int count)
{
    int t = blockIdx.x * blockDim.x + threadIdx.x;
    if (t < count) a[t] = fmaxf(a[t] + x[t], 0.f);
}

// fi[b,c,m] *= gate[b,0,m]
__global__ void gate_mul_kernel(float* __restrict__ fi, const float* __restrict__ gate,
                                int C, int M)
{
    int t = blockIdx.x * blockDim.x + threadIdx.x;
    int total = 2 * C * M;
    if (t >= total) return;
    int b = t / (C * M);
    int m = t % M;
    fi[t] *= gate[(size_t)b * M + m];
}

// dst[b, coff+c, n] = src[b, c, n]  (dst has Ctot channels)
__global__ void copy_chan_kernel(float* __restrict__ dst, const float* __restrict__ src,
                                 int C, int N, int Ctot, int coff)
{
    int t = blockIdx.x * blockDim.x + threadIdx.x;
    int total = 2 * C * N;
    if (t >= total) return;
    int b = t / (C * N);
    int r = t % (C * N);
    int c = r / N, n = r % N;
    dst[((size_t)b * Ctot + coff + c) * N + n] = src[((size_t)b * C + c) * N + n];
}

// In-place softmax along N for each (b,k) row; grid = B*Ck blocks.
__global__ __launch_bounds__(256)
void softmax_rows_kernel(float* __restrict__ a, int N)
{
    float* p = a + (size_t)blockIdx.x * N;
    __shared__ float red[8];
    const int tid = threadIdx.x;
    const int lane = tid & 31, wv = tid >> 5;

    float m = -INFINITY;
    for (int i = tid; i < N; i += 256) m = fmaxf(m, p[i]);
    for (int off = 16; off; off >>= 1) m = fmaxf(m, __shfl_xor(m, off));
    if (lane == 0) red[wv] = m;
    __syncthreads();
    if (tid == 0) {
        float v = red[0];
        for (int i = 1; i < 8; ++i) v = fmaxf(v, red[i]);
        red[0] = v;
    }
    __syncthreads();
    const float mx = red[0];
    __syncthreads();

    float s = 0.f;
    for (int i = tid; i < N; i += 256) {
        float e = __expf(p[i] - mx);
        p[i] = e;
        s += e;
    }
    for (int off = 16; off; off >>= 1) s += __shfl_xor(s, off);
    if (lane == 0) red[wv] = s;
    __syncthreads();
    if (tid == 0) {
        float v = 0.f;
        for (int i = 0; i < 8; ++i) v += red[i];
        red[0] = v;
    }
    __syncthreads();
    const float inv = 1.f / red[0];
    for (int i = tid; i < N; i += 256) p[i] *= inv;
}

// a[b,:,n] /= (1e-9 + sum_c a[b,c,n])
__global__ void colnorm_kernel(float* __restrict__ a, int Ck, int N)
{
    int t = blockIdx.x * blockDim.x + threadIdx.x;
    if (t >= 2 * N) return;
    int b = t / N, n = t % N;
    float* base = a + (size_t)b * Ck * N + n;
    float s = 0.f;
    for (int c = 0; c < Ck; ++c) s += base[(size_t)c * N];
    float inv = 1.f / (1e-9f + s);
    for (int c = 0; c < Ck; ++c) base[(size_t)c * N] *= inv;
}

// ---------------------------------------------------------------------------
// Host orchestration
// ---------------------------------------------------------------------------
extern "C" void kernel_launch(void* const* d_in, const int* in_sizes, int n_in,
                              void* d_out, int out_size, void* d_ws, size_t ws_size,
                              hipStream_t stream)
{
    (void)in_sizes; (void)n_in; (void)out_size; (void)ws_size;
    const int NP[5] = {40960, 10240, 2560, 640, 160};

    // ---- input mapping: top-level in insertion order; 'params' expanded as
    //      jax tree_flatten leaves (dict keys sorted alphabetically). ----
    const float* features = (const float*)d_in[0];
    int cur = 1;
    auto F = [&]() { return (const float*)d_in[cur++]; };

    const float *all_b[NL], *all_g[NL], *all_w[NL];
    for (int j = 0; j < NL; ++j) { all_b[j] = F(); all_g[j] = F(); all_w[j] = F(); }
    const float *at_b1 = F(), *at_bp = F(), *at_bq = F();
    const float *at_w0 = F(), *at_w1 = F(), *at_wp = F(), *at_wq = F();
    const float *dec_b[NL], *dec_g[NL], *dec_w[NL];
    for (int j = 0; j < NL; ++j) { dec_b[j] = F(); dec_g[j] = F(); dec_w[j] = F(); }
    const float *dec0_b = F(), *dec0_g = F(), *dec0_w = F();
    struct EncP { const float *pe_b, *pe_g, *pe_w, *m1b, *m1g, *m1w, *m2b, *m2g, *m2w, *sb, *sg, *sw; } ep[NL];
    for (int i = 0; i < NL; ++i) {
        ep[i].pe_b = F(); ep[i].pe_g = F(); ep[i].pe_w = F();
        ep[i].m1b = F(); ep[i].m1g = F(); ep[i].m1w = F();
        ep[i].m2b = F(); ep[i].m2g = F(); ep[i].m2w = F();
        ep[i].sb = F(); ep[i].sg = F(); ep[i].sw = F();
    }
    const float *fc0_b = F(), *fc0_be = F(), *fc0_g = F(), *fc0_w = F();
    const float *fc1_b = F(), *fc1_g = F(), *fc1_w = F();
    const float *fc2_b = F(), *fc2_g = F(), *fc2_w = F();
    const float *fc3_b = F(), *fc3_w = F();
    const float *mlp_b[NL], *mlp_g[NL], *mlp_w[NL];
    for (int j = 0; j < NL; ++j) { mlp_b[j] = F(); mlp_g[j] = F(); mlp_w[j] = F(); }

    const float* xyz[NL];
    const int *neigh[NL], *sub[NL], *interp[NL];
    for (int i = 0; i < NL; ++i) {
        xyz[i]    = (const float*)d_in[cur++];
        neigh[i]  = (const int*)d_in[cur++];
        sub[i]    = (const int*)d_in[cur++];
        interp[i] = (const int*)d_in[cur++];
    }

    // ---- workspace bump allocator ----
    float* wsf = (float*)d_ws;
    size_t off = 0;
    auto alloc = [&](size_t elems) {
        float* p = wsf + off;
        off += (elems + 63) & ~(size_t)63;
        return p;
    };
    const size_t ARENA = (size_t)2 * 64 * 40960;  // max single tensor: [2,64,40960]
    float* a0 = alloc(ARENA);
    float* a1 = alloc(ARENA);
    float* a2 = alloc(ARENA);
    float* a3 = alloc(ARENA);
    float* fe0 = alloc((size_t)2 * 32 * NP[0]);
    float* fsub[NL];
    const int d2[NL] = {32, 128, 256, 512};       // 2*d per level
    for (int i = 0; i < NL; ++i) fsub[i] = alloc((size_t)2 * d2[i] * NP[i + 1]);

    // ---- launch helpers ----
    auto gemm = [&](const float* X, const float* W, const float* g, const float* be,
                    const float* R, float* Y, int Cin, int Cout, int N, int act) {
        dim3 grid((N + BN - 1) / BN, (Cout + BM - 1) / BM, 2);
        gemm_bn_act_kernel<<<grid, 256, 0, stream>>>(X, W, g, be, R, Y, Cin, Cout, N, act);
    };
    auto ew = [](int n) { return dim3((n + 255) / 256); };

    // ---- fc0 ----
    fc0_kernel<<<ew(2 * NP[0]), 256, 0, stream>>>(features, fc0_w, fc0_b, fc0_g, fc0_be, a0, NP[0]);

    // ---- encoder ----
    const int dh[NL] = {8, 32, 64, 128};   // d/2
    const float* fin = a0;
    int cin = 8;
    for (int i = 0; i < NL; ++i) {
        int N = NP[i], Nn = NP[i + 1], ch = dh[i], c2 = d2[i];
        float *h1, *lf, *sc, *fe;
        if (i == 0) { h1 = a1; lf = a2; sc = a3; fe = fe0; }
        else        { h1 = a0; lf = a1; sc = a2; fe = a3; }
        // h = relu(mlp1(f))
        gemm(fin, ep[i].m1w, ep[i].m1g, ep[i].m1b, nullptr, h1, cin, ch, N, ACT_RELU);
        // h = lfa(xyz, h, neigh)  -> [B, 2*ch, N]
        lfa_kernel<<<dim3(N, 2), 64, 0, stream>>>(xyz[i], h1, neigh[i],
                                                  ep[i].pe_w, ep[i].pe_g, ep[i].pe_b, lf, ch, N);
        // sc = short(f) (no act)
        gemm(fin, ep[i].sw, ep[i].sg, ep[i].sb, nullptr, sc, cin, c2, N, ACT_NONE);
        // fe = leaky_0.2(mlp2(h) + sc)
        gemm(lf, ep[i].m2w, ep[i].m2g, ep[i].m2b, sc, fe, 2 * ch, c2, N, ACT_LEAK2);
        // f = max-pool subsample
        pool_max_kernel<<<ew(2 * c2 * Nn), 256, 0, stream>>>(fe, sub[i], fsub[i], c2, N, Nn);
        fin = fsub[i];
        cin = c2;
    }

    // ---- decoder ----
    // f = relu(dec0(fsub3)) : 512 -> 512 @ N=160
    gemm(fsub[3], dec0_w, dec0_g, dec0_b, nullptr, a0, 512, 512, NP[4], ACT_RELU);

    const float* skp[NL] = {fsub[2], fsub[1], fsub[0], fe0};
    const int skC[NL]   = {256, 128, 32, 32};
    const int dU[NL]    = {512, 256, 128, 32};
    const int dIn[NL]   = {768, 384, 160, 64};
    const int dOutC[NL] = {256, 128, 32, 32};
    for (int j = 0; j < NL; ++j) {
        int M = NP[3 - j], Nsrc = NP[4 - j], du = dU[j], cs = skC[j];
        // fi = gather(f, interp)
        gather_kernel<<<ew(2 * du * M), 256, 0, stream>>>(a0, interp[3 - j], a1, du, Nsrc, M);
        // adj = mlp_j(skip)  (no act)
        gemm(skp[j], mlp_w[j], mlp_g[j], mlp_b[j], nullptr, a2, cs, du, M, ACT_NONE);
        // t = relu(adj + fi)  (in-place into a2)
        add_relu_kernel<<<ew(2 * du * M), 256, 0, stream>>>(a2, a1, 2 * du * M);
        // gate = sigmoid(all_j(t)) : du -> 1
        gemm(a2, all_w[j], all_g[j], all_b[j], nullptr, a3, du, 1, M, ACT_SIGM);
        // fi *= gate
        gate_mul_kernel<<<ew(2 * du * M), 256, 0, stream>>>(a1, a3, du, M);
        // concat([skip, fi]) into a2
        copy_chan_kernel<<<ew(2 * cs * M), 256, 0, stream>>>(a2, skp[j], cs, M, dIn[j], 0);
        copy_chan_kernel<<<ew(2 * du * M), 256, 0, stream>>>(a2, a1, du, M, dIn[j], cs);
        // f = relu(dec_j(concat))
        gemm(a2, dec_w[j], dec_g[j], dec_b[j], nullptr, a0, dIn[j], dOutC[j], M, ACT_RELU);
    }

    // ---- external attention (C = 32, N = 40960, channel-major throughout) ----
    const int N0 = NP[0];
    // xq = wq*f + bq   (this is "idn")
    gemm(a0, at_wq, nullptr, at_bq, nullptr, a1, 32, 32, N0, ACT_NONE);
    // attn logits: a = w0 * xq  -> [B,64,N]
    gemm(a1, at_w0, nullptr, nullptr, nullptr, a2, 32, 64, N0, ACT_NONE);
    // softmax over N per (b,k), then double-normalize over k per (b,n)
    softmax_rows_kernel<<<dim3(2 * 64), 256, 0, stream>>>(a2, N0);
    colnorm_kernel<<<ew(2 * N0), 256, 0, stream>>>(a2, 64, N0);
    // x = w1 * a + b1  -> [B,32,N]
    gemm(a2, at_w1, nullptr, at_b1, nullptr, a3, 64, 32, N0, ACT_NONE);
    // out = idn + (wp * x + bp)
    gemm(a3, at_wp, nullptr, at_bp, a1, a2, 32, 32, N0, ACT_NONE);

    // ---- head ----
    gemm(a2, fc1_w, fc1_g, fc1_b, nullptr, a3, 32, 64, N0, ACT_RELU);
    gemm(a3, fc2_w, fc2_g, fc2_b, nullptr, a1, 64, 32, N0, ACT_RELU);
    gemm(a1, fc3_w, nullptr, fc3_b, nullptr, (float*)d_out, 32, 13, N0, ACT_NONE);
}
